// ALiBiMultiHeadAttention_26989574488462
// MI455X (gfx1250) — matrix operand order repaired and checked by reading, not compiled, not run on hardware
//
#include <hip/hip_runtime.h>
#include <hip/hip_bf16.h>

// ---------------- CDNA5 WMMA types ----------------
typedef __attribute__((ext_vector_type(16))) __bf16 v16bf;
typedef __attribute__((ext_vector_type(8)))  float  v8f;

union BF16x16 {
    unsigned int u[8];
    v16bf v;
};

__device__ __forceinline__ unsigned short f2bf(float f) {
    unsigned int u = __float_as_uint(f);
    unsigned int r = u + 0x7FFFu + ((u >> 16) & 1u);   // RNE
    return (unsigned short)(r >> 16);
}

#define S_LEN 2048
#define NHEAD 16
#define HD 64
#define DM 1024

// =====================================================================
// Kernel 1: kqv = x @ W_kqv, scattered to K/Q/V [B][H][S][64] as bf16
// Block: 256 thr (8 waves). Tile: M=128 (16 rows/wave), N=64 (4 wmma cols)
// =====================================================================
__global__ __launch_bounds__(256) void k_qkv(const float* __restrict__ x,
                                             const float* __restrict__ w,
                                             unsigned short* __restrict__ Kw,
                                             unsigned short* __restrict__ Qw,
                                             unsigned short* __restrict__ Vw) {
    __shared__ alignas(16) unsigned short As[128 * 36];  // [row][k] padded
    __shared__ alignas(16) unsigned short Bs[64 * 36];   // [n][k]  transposed, padded

    const int tid  = threadIdx.x;
    const int wave = tid >> 5;
    const int lane = tid & 31;
    const int lo16 = lane & 15;
    const int hi   = (lane >= 16) ? 1 : 0;
    const int m0 = blockIdx.y * 128;
    const int n0 = blockIdx.x * 64;

    v8f acc[4] = {};

    for (int k0 = 0; k0 < DM; k0 += 32) {
        __syncthreads();
        // stage A: 128x32 fp32 -> bf16, float2 per load, 8 per thread
        #pragma unroll
        for (int i = 0; i < 8; ++i) {
            int idx = tid + i * 256;            // 0..2047
            int r   = idx >> 4;                 // 0..127
            int c2  = idx & 15;                 // float2 column
            float2 f = *(const float2*)(x + (size_t)(m0 + r) * DM + k0 + c2 * 2);
            unsigned int packed = (unsigned int)f2bf(f.x) | ((unsigned int)f2bf(f.y) << 16);
            *(unsigned int*)&As[r * 36 + c2 * 2] = packed;
        }
        // stage B transposed: W[k0+k][n0+n] -> Bs[n*36 + k], 4 float2 per thread
        #pragma unroll
        for (int i = 0; i < 4; ++i) {
            int idx = tid + i * 256;            // 0..1023
            int kk  = idx >> 5;                 // 0..31
            int n2  = idx & 31;                 // pair of n
            float2 f = *(const float2*)(w + (size_t)(k0 + kk) * 3072 + n0 + n2 * 2);
            Bs[(n2 * 2 + 0) * 36 + kk] = f2bf(f.x);
            Bs[(n2 * 2 + 1) * 36 + kk] = f2bf(f.y);
        }
        __syncthreads();

        // A fragment: lane=row (lo16), contiguous K run of 16 at 16*hi
        BF16x16 a;
        {
            const unsigned short* ap = &As[((wave << 4) + lo16) * 36];
            int koff = hi * 16;
            #pragma unroll
            for (int j = 0; j < 4; ++j) {
                a.u[j]     = *(const unsigned int*)(ap + koff + 2 * j);
                a.u[4 + j] = *(const unsigned int*)(ap + 8 + koff + 2 * j);
            }
        }
        // 4 column tiles: B lane=col, contiguous K run of 16
        #pragma unroll
        for (int c = 0; c < 4; ++c) {
            BF16x16 bfr;
            const unsigned short* bp = &Bs[((c << 4) + lo16) * 36 + hi * 16];
            #pragma unroll
            for (int j = 0; j < 4; ++j) {
                uint2 t = *(const uint2*)(bp + 4 * j);
                bfr.u[2 * j] = t.x; bfr.u[2 * j + 1] = t.y;
            }
            acc[c] = __builtin_amdgcn_wmma_f32_16x16x32_bf16(
                false, a.v, false, bfr.v, (short)0, acc[c], false, false);
        }
    }

    // scatter: n0 aligned to 64 -> one (chunk, head) per block
    const int chunk = n0 >> 10;                 // 0=K,1=Q,2=V
    const int head  = (n0 & 1023) >> 6;
    unsigned short* dst = (chunk == 0) ? Kw : ((chunk == 1) ? Qw : Vw);
    #pragma unroll
    for (int c = 0; c < 4; ++c) {
        int dim = (c << 4) + lo16;
        #pragma unroll
        for (int r = 0; r < 8; ++r) {
            int mg = m0 + (wave << 4) + r + hi * 8;     // global row = b*S + s
            int bb = mg >> 11;
            int ss = mg & 2047;
            size_t off = ((size_t)((bb * NHEAD + head) * S_LEN + ss)) * HD + dim;
            dst[off] = f2bf(acc[c][r]);
        }
    }
}

// =====================================================================
// Kernel 2: causal ALiBi flash attention, bf16 WMMA, fp32 online softmax
// Grid: (S/128, B*H). Block 256 = 8 waves; each wave owns 16 query rows.
// =====================================================================
__global__ __launch_bounds__(256) void k_attn(const unsigned short* __restrict__ Kw,
                                              const unsigned short* __restrict__ Qw,
                                              const unsigned short* __restrict__ Vw,
                                              const float* __restrict__ slopes,
                                              float* __restrict__ out) {
    __shared__ alignas(16) unsigned short Vt[64 * 36];       // [d][k] transposed V tile
    __shared__ alignas(16) unsigned short Ps[8][16 * 36];    // per-wave P tile [row][k]

    const int tid  = threadIdx.x;
    const int wave = tid >> 5;
    const int lane = tid & 31;
    const int lo16 = lane & 15;
    const int hi   = (lane >= 16) ? 1 : 0;

    const int qt = blockIdx.x;
    const int bh = blockIdx.y;
    const int bb = bh >> 4;
    const int hh = bh & 15;
    const float slope = slopes[hh];
    const size_t base = (size_t)bh * S_LEN * HD;
    const int q0 = qt * 128 + wave * 16;

    // ---- load Q fragments (d 0..31 and 32..63) straight from global ----
    BF16x16 qf[2];
    {
        const unsigned short* qp = Qw + base + (size_t)(q0 + lo16) * HD;
        int koff = hi * 16;
        #pragma unroll
        for (int f = 0; f < 2; ++f) {
            int d0 = f * 32;
            #pragma unroll
            for (int j = 0; j < 4; ++j) {
                qf[f].u[j]     = *(const unsigned int*)(qp + d0 + koff + 2 * j);
                qf[f].u[4 + j] = *(const unsigned int*)(qp + d0 + 8 + koff + 2 * j);
            }
        }
    }

    v8f oacc[4] = {};
    float mrow[8], lrow[8];
    #pragma unroll
    for (int r = 0; r < 8; ++r) { mrow[r] = -1e30f; lrow[r] = 0.0f; }

    const int nkeys = qt * 128 + 128;
    for (int kb = 0; kb < nkeys; kb += 32) {
        __syncthreads();
        // ---- cooperative stage of V tile transposed into LDS ----
        #pragma unroll
        for (int i = 0; i < 2; ++i) {
            int idx = tid + i * 256;        // 0..511 groups of 4 bf16
            int k   = idx >> 4;             // 0..31
            int d4  = (idx & 15) * 4;       // 0..60
            uint2 t = *(const uint2*)(Vw + base + (size_t)(kb + k) * HD + d4);
            Vt[(d4 + 0) * 36 + k] = (unsigned short)(t.x & 0xffffu);
            Vt[(d4 + 1) * 36 + k] = (unsigned short)(t.x >> 16);
            Vt[(d4 + 2) * 36 + k] = (unsigned short)(t.y & 0xffffu);
            Vt[(d4 + 3) * 36 + k] = (unsigned short)(t.y >> 16);
        }
        __syncthreads();

        // ---- S = Q K^T : two 16-key tiles, K fragments direct from global ----
        v8f sa[2];
        #pragma unroll
        for (int c = 0; c < 2; ++c) {
            const unsigned short* kp =
                Kw + base + (size_t)(kb + c * 16 + lo16) * HD + hi * 16;
            BF16x16 b0, b1;
            *(uint4*)&b0.u[0] = *(const uint4*)(kp);
            *(uint4*)&b0.u[4] = *(const uint4*)(kp + 8);
            *(uint4*)&b1.u[0] = *(const uint4*)(kp + 32);
            *(uint4*)&b1.u[4] = *(const uint4*)(kp + 40);
            v8f z = {};
            z = __builtin_amdgcn_wmma_f32_16x16x32_bf16(
                false, qf[0].v, false, b0.v, (short)0, z, false, false);
            z = __builtin_amdgcn_wmma_f32_16x16x32_bf16(
                false, qf[1].v, false, b1.v, (short)0, z, false, false);
            sa[c] = z;
        }

        // ---- scale + ALiBi bias + causal mask + online softmax ----
        #pragma unroll
        for (int r = 0; r < 8; ++r) {
            int qi = q0 + r + hi * 8;
            float sv0, sv1;
            {
                int j0 = kb + lo16;
                int j1 = kb + 16 + lo16;
                sv0 = sa[0][r] * 0.03125f + slope * (float)(j0 - qi);
                sv1 = sa[1][r] * 0.03125f + slope * (float)(j1 - qi);
                sv0 = (j0 <= qi) ? sv0 : -1e30f;
                sv1 = (j1 <= qi) ? sv1 : -1e30f;
            }
            float tmax = fmaxf(sv0, sv1);
            #pragma unroll
            for (int d = 1; d < 16; d <<= 1)
                tmax = fmaxf(tmax, __shfl_xor(tmax, d, 32));
            float mnew  = fmaxf(mrow[r], tmax);
            float alpha = __expf(mrow[r] - mnew);
            float p0 = __expf(sv0 - mnew);
            float p1 = __expf(sv1 - mnew);
            float psum = p0 + p1;
            #pragma unroll
            for (int d = 1; d < 16; d <<= 1)
                psum += __shfl_xor(psum, d, 32);
            lrow[r] = lrow[r] * alpha + psum;
            mrow[r] = mnew;
            #pragma unroll
            for (int dt = 0; dt < 4; ++dt) oacc[dt][r] *= alpha;
            // C-layout -> LDS row-major P (per-wave region)
            unsigned short* pr = &Ps[wave][(r + hi * 8) * 36];
            pr[lo16]      = f2bf(p0);
            pr[16 + lo16] = f2bf(p1);
        }

        // ---- O += P @ V ----
        BF16x16 pa;
        {
            const unsigned short* pp = &Ps[wave][lo16 * 36];
            int koff = hi * 16;
            #pragma unroll
            for (int j = 0; j < 4; ++j) {
                pa.u[j]     = *(const unsigned int*)(pp + koff + 2 * j);
                pa.u[4 + j] = *(const unsigned int*)(pp + 8 + koff + 2 * j);
            }
        }
        #pragma unroll
        for (int dt = 0; dt < 4; ++dt) {
            const unsigned short* vp = &Vt[((dt << 4) + lo16) * 36 + hi * 16];
            BF16x16 vb;
            #pragma unroll
            for (int j = 0; j < 4; ++j) {
                uint2 t = *(const uint2*)(vp + 4 * j);
                vb.u[2 * j] = t.x; vb.u[2 * j + 1] = t.y;
            }
            oacc[dt] = __builtin_amdgcn_wmma_f32_16x16x32_bf16(
                false, pa.v, false, vb.v, (short)0, oacc[dt], false, false);
        }
    }

    // ---- normalize and write out [B][S][H][64] fp32 ----
    #pragma unroll
    for (int r = 0; r < 8; ++r) {
        float inv = 1.0f / lrow[r];
        int ss = q0 + r + hi * 8;
        #pragma unroll
        for (int dt = 0; dt < 4; ++dt) {
            int d = (dt << 4) + lo16;
            out[(((size_t)bb * S_LEN + ss) * NHEAD + hh) * HD + d] = oacc[dt][r] * inv;
        }
    }
}

// =====================================================================
extern "C" void kernel_launch(void* const* d_in, const int* in_sizes, int n_in,
                              void* d_out, int out_size, void* d_ws, size_t ws_size,
                              hipStream_t stream) {
    const float* x = (const float*)d_in[0];      // [2, 2048, 1024]
    const float* w = (const float*)d_in[1];      // [1024, 3072]
    const float* m = (const float*)d_in[2];      // [16]
    float* out = (float*)d_out;                  // [2, 2048, 1024]

    // workspace: K, Q, V each [2][16][2048][64] bf16 = 4194304 elems
    unsigned short* Kw = (unsigned short*)d_ws;
    unsigned short* Qw = Kw + 4194304;
    unsigned short* Vw = Kw + 8388608;

    dim3 g1(3072 / 64, 4096 / 128);   // (48, 32)
    k_qkv<<<g1, dim3(256), 0, stream>>>(x, w, Kw, Qw, Vw);

    dim3 g2(S_LEN / 128, 2 * NHEAD);  // (16, 32)
    k_attn<<<g2, dim3(256), 0, stream>>>(Kw, Qw, Vw, m, out);
}
